// ZoomTransformerBlock_74302934221438
// MI455X (gfx1250) — matrix operand order repaired and checked by reading, not compiled, not run on hardware
//
#include <hip/hip_runtime.h>
#include <hip/hip_bf16.h>

// ---------------- types ----------------
typedef __bf16  v16bf __attribute__((ext_vector_type(16)));
typedef float   v8f   __attribute__((ext_vector_type(8)));
typedef unsigned int v4u __attribute__((ext_vector_type(4)));
typedef unsigned int v8u __attribute__((ext_vector_type(8)));
typedef __hip_bfloat16 bf16s;

// ---------------- problem constants ----------------
constexpr int   Bc   = 2;
constexpr int   Nn   = 2048;
constexpr int   Dd   = 1024;
constexpr int   Hh   = 16;
constexpr int   DHh  = 64;
constexpr int   D3   = 3072;
constexpr int   DF   = 4096;
constexpr int   Mrows = Bc * Nn;           // 4096
constexpr float SCALE_F = 0.03125f;        // 1024^-0.5

// ---------------- helpers ----------------
__device__ __forceinline__ v16bf combine16(v4u a, v4u b) {
  v8u w = __builtin_shufflevector(a, b, 0, 1, 2, 3, 4, 5, 6, 7);
  return __builtin_bit_cast(v16bf, w);
}
__device__ __forceinline__ v16bf ld_frag32(const bf16s* p) {  // 32B contiguous
  return combine16(*(const v4u*)p, *(const v4u*)(p + 8));
}
__device__ __forceinline__ v8f wmma_bf16(v16bf a, v16bf b, v8f c) {
  return __builtin_amdgcn_wmma_f32_16x16x32_bf16(false, a, false, b, (short)0, c,
                                                 false, false);
}
__device__ __forceinline__ float redmax16(float v) {
#pragma unroll
  for (int m = 1; m < 16; m <<= 1) v = fmaxf(v, __shfl_xor(v, m, 32));
  return v;
}
__device__ __forceinline__ float redsum16(float v) {
#pragma unroll
  for (int m = 1; m < 16; m <<= 1) v += __shfl_xor(v, m, 32);
  return v;
}
// async copy of 16B global -> LDS, tracked by ASYNCcnt (CDNA5 path)
__device__ __forceinline__ void async_g2l_b128(const bf16s* g, bf16s* l) {
  unsigned lds_off = (unsigned)(unsigned long long)(const void*)l;
  unsigned long long ga = (unsigned long long)(const void*)g;
  asm volatile("global_load_async_to_lds_b128 %0, %1, off"
               :: "v"(lds_off), "v"(ga) : "memory");
}
__device__ __forceinline__ void wait_async0() {
  asm volatile("s_wait_asynccnt 0x0" ::: "memory");
}

// ---------------- weight convert + transpose: wt[n][k] = (bf16)w[k][n] ----
__global__ void wt_convert_kernel(const float* __restrict__ w,
                                  bf16s* __restrict__ wt, int K, int Nc) {
  size_t idx = (size_t)blockIdx.x * 256 + threadIdx.x;
  size_t total = (size_t)K * Nc;
  if (idx >= total) return;
  int n = (int)(idx / K);
  int k = (int)(idx % K);
  wt[idx] = __float2bfloat16(w[(size_t)k * Nc + n]);
}

// ---------------- LayerNorm (f32 in -> bf16 out) ----------------
__global__ __launch_bounds__(256) void ln_kernel(const float* __restrict__ x,
                                                 const float* __restrict__ w,
                                                 const float* __restrict__ b,
                                                 bf16s* __restrict__ out) {
  int row = blockIdx.x;
  const float* xr = x + (size_t)row * Dd;
  float s = 0.f, s2 = 0.f;
  for (int i = threadIdx.x; i < Dd; i += 256) {
    float v = xr[i];
    s += v;
    s2 += v * v;
  }
#pragma unroll
  for (int m = 1; m < 32; m <<= 1) {
    s += __shfl_xor(s, m, 32);
    s2 += __shfl_xor(s2, m, 32);
  }
  __shared__ float ss[8], ss2[8];
  if ((threadIdx.x & 31) == 0) {
    ss[threadIdx.x >> 5] = s;
    ss2[threadIdx.x >> 5] = s2;
  }
  __syncthreads();
  s = 0.f; s2 = 0.f;
#pragma unroll
  for (int i = 0; i < 8; i++) { s += ss[i]; s2 += ss2[i]; }
  float mu = s * (1.f / Dd);
  float var = s2 * (1.f / Dd) - mu * mu;
  float rstd = rsqrtf(var + 1e-5f);
  for (int i = threadIdx.x; i < Dd; i += 256) {
    float v = (xr[i] - mu) * rstd * w[i] + b[i];
    out[(size_t)row * Dd + i] = __float2bfloat16(v);
  }
}

// ---------------- generic bf16 WMMA GEMM with async-LDS W staging ----------
// C[M,Nc] = A[M,K](bf16) * W[K,Nc] (via Wt[Nc][K] bf16) + bias
// block = 8 waves stacked in M -> 128x64 tile; the shared 64x64 Wt slab is
// staged once per block via global_load_async_to_lds_b128, double-buffered.
template <bool OUT_BF16, bool DO_GELU, bool DO_RESID, bool QKV_MODE>
__global__ __launch_bounds__(256) void gemm_wmma_kernel(
    const bf16s* __restrict__ A, const bf16s* __restrict__ Wt,
    const float* __restrict__ bias, const float* __restrict__ resid,
    void* __restrict__ outp, bf16s* __restrict__ vtp, int M, int Nc, int K) {
  __shared__ alignas(64) bf16s sW[2][64 * 64];  // [buf][col][64k] = 8KB each

  int tid = threadIdx.x;
  int lane = tid & 31;
  int w = tid >> 5;
  int l16 = lane & 15;
  int hi = lane >> 4;  // 0 | 1
  int mb = blockIdx.y * 128 + w * 16;
  int nb = blockIdx.x * 64;

  // cooperative W-slab copy: 64 rows x 128B; 4 threads/row, 2x16B each
  int wrow = tid >> 2;          // 0..63
  int wch = (tid & 3) * 16;     // bf16 offset within row (2 chunks of 8)
  const bf16s* wsrc = Wt + (size_t)(nb + wrow) * K + wch;

  v8f acc[4];
#pragma unroll
  for (int t = 0; t < 4; t++) acc[t] = (v8f){0, 0, 0, 0, 0, 0, 0, 0};

  // prologue: ks=0 slab -> buf0
  async_g2l_b128(wsrc, &sW[0][wrow * 64 + wch]);
  async_g2l_b128(wsrc + 8, &sW[0][wrow * 64 + wch + 8]);

  const bf16s* arow = A + (size_t)(mb + l16) * K;
  for (int ks = 0; ks < K; ks += 64) {
    int ib = (ks >> 6) & 1;
    wait_async0();
    __syncthreads();
    if (ks + 64 < K) {
      async_g2l_b128(wsrc + ks + 64, &sW[ib ^ 1][wrow * 64 + wch]);
      async_g2l_b128(wsrc + ks + 72, &sW[ib ^ 1][wrow * 64 + wch + 8]);
    }

    // A fragments for the two 32-wide sub-steps
    v16bf av[2];
#pragma unroll
    for (int s = 0; s < 2; s++)
      av[s] = combine16(*(const v4u*)(arow + ks + s * 32 + hi * 16),
                        *(const v4u*)(arow + ks + s * 32 + hi * 16 + 8));
    // preload all 8 B fragments (distinct regs -> DS/WMMA overlap)
    v16bf bv[2][4];
#pragma unroll
    for (int s = 0; s < 2; s++)
#pragma unroll
      for (int t = 0; t < 4; t++)
        bv[s][t] = ld_frag32(&sW[ib][(t * 16 + l16) * 64 + s * 32 + hi * 16]);
#pragma unroll
    for (int s = 0; s < 2; s++)
#pragma unroll
      for (int t = 0; t < 4; t++) acc[t] = wmma_bf16(av[s], bv[s][t], acc[t]);
  }

#pragma unroll
  for (int t = 0; t < 4; t++) {
    int col = nb + t * 16 + l16;
    float bv = bias[col];
#pragma unroll
    for (int r = 0; r < 8; r++) {
      int row = mb + r + hi * 8;
      float v = acc[t][r] + bv;
      if constexpr (DO_GELU) v = 0.5f * v * (1.f + erff(v * 0.70710678118654752f));
      if constexpr (DO_RESID) v += resid[(size_t)row * Nc + col];
      if constexpr (OUT_BF16)
        ((bf16s*)outp)[(size_t)row * Nc + col] = __float2bfloat16(v);
      else
        ((float*)outp)[(size_t)row * Nc + col] = v;
      if constexpr (QKV_MODE) {
        if (col >= 2 * Dd) {  // V columns -> also emit V^T [b][h][d][token]
          int hd = col - 2 * Dd;
          int h = hd >> 6, d = hd & 63;
          int bb = row >> 11, tok = row & (Nn - 1);
          vtp[((size_t)((bb * Hh + h) * DHh + d)) * Nn + tok] = __float2bfloat16(v);
        }
      }
    }
  }
}

// ---------------- flash attention ----------------
// One wave = 16 query rows of one head; all 8 waves of a block share (b,h),
// so the 32-key K tile and V^T tile are staged once per block into LDS via
// async copies, double-buffered across the key loop.
__global__ __launch_bounds__(256) void flash_attn_kernel(
    const bf16s* __restrict__ qkv, const bf16s* __restrict__ vt,
    const float* __restrict__ gamma, const float* __restrict__ dist,
    bf16s* __restrict__ attn_out) {
  __shared__ alignas(64) bf16s sK[2][32 * 64];  // [buf][key][64d] 4KB
  __shared__ alignas(64) bf16s sV[2][64 * 32];  // [buf][d][32key] 4KB
  __shared__ alignas(64) bf16s sP[8][16 * 32];  // per-wave P tile

  int tid = threadIdx.x;
  int lane = tid & 31;
  int w = tid >> 5;
  int l16 = lane & 15;
  int hi = lane >> 4;
  int gw = blockIdx.x * 8 + w;     // 0..4095
  int qt = gw & 127;               // query tile (differs per wave)
  int h = (gw >> 7) & 15;          // same for all 8 waves in block
  int b = gw >> 11;                // same for all 8 waves in block
  int qb = qt * 16;
  const float g = gamma[b * Hh + h];

  // per-thread slices of the cooperative K / V^T tile copies (16B each)
  int krow = tid >> 3;             // 0..31 (key)
  int kch = (tid & 7) * 8;         // dim offset
  const bf16s* ksrc = qkv + (size_t)(b * Nn + krow) * D3 + Dd + h * DHh + kch;
  int vrow = tid >> 2;             // 0..63 (dim)
  int vch = (tid & 3) * 8;         // key offset
  const bf16s* vsrc = vt + ((size_t)((b * Hh + h) * DHh + vrow)) * Nn + vch;

  // Q fragments for the two K-steps of DH=64
  v16bf qa[2];
  const bf16s* qrow = qkv + (size_t)(b * Nn + qb + l16) * D3 + h * DHh;
#pragma unroll
  for (int s = 0; s < 2; s++)
    qa[s] = combine16(*(const v4u*)(qrow + s * 32 + hi * 16),
                      *(const v4u*)(qrow + s * 32 + hi * 16 + 8));

  float mrow[8], lrow[8];
  v8f acc[4];
#pragma unroll
  for (int r = 0; r < 8; r++) { mrow[r] = -1e30f; lrow[r] = 0.f; }
#pragma unroll
  for (int t = 0; t < 4; t++) acc[t] = (v8f){0, 0, 0, 0, 0, 0, 0, 0};

  bf16s* lds = &sP[w][0];

  // prologue: stage kb=0 tiles into buffer 0
  async_g2l_b128(ksrc, &sK[0][krow * 64 + kch]);
  async_g2l_b128(vsrc, &sV[0][vrow * 32 + vch]);

  for (int kb = 0; kb < Nn; kb += 32) {
    int ib = (kb >> 5) & 1;
    wait_async0();
    __syncthreads();
    if (kb + 32 < Nn) {
      async_g2l_b128(ksrc + (size_t)(kb + 32) * D3, &sK[ib ^ 1][krow * 64 + kch]);
      async_g2l_b128(vsrc + (kb + 32), &sV[ib ^ 1][vrow * 32 + vch]);
    }

    // ---- S = Q K^T: preload 4 K fragments, then chain WMMAs ----
    v16bf kf[2][2];  // [keytile][substep]
#pragma unroll
    for (int t = 0; t < 2; t++)
#pragma unroll
      for (int s = 0; s < 2; s++)
        kf[t][s] = ld_frag32(&sK[ib][(t * 16 + l16) * 64 + s * 32 + hi * 16]);
    v8f S0 = (v8f){0, 0, 0, 0, 0, 0, 0, 0};
    v8f S1 = (v8f){0, 0, 0, 0, 0, 0, 0, 0};
    S0 = wmma_bf16(qa[0], kf[0][0], S0);
    S1 = wmma_bf16(qa[0], kf[1][0], S1);
    S0 = wmma_bf16(qa[1], kf[0][1], S0);
    S1 = wmma_bf16(qa[1], kf[1][1], S1);

    // ---- online softmax ----
    int key0 = kb + l16, key1 = kb + 16 + l16;
    float p0[8], p1[8], corr[8];
#pragma unroll
    for (int r = 0; r < 8; r++) {
      int qi = qb + r + hi * 8;
      float s0 = S0[r] * SCALE_F - g * dist[(size_t)qi * Nn + key0];
      float s1 = S1[r] * SCALE_F - g * dist[(size_t)qi * Nn + key1];
      float mx = redmax16(fmaxf(s0, s1));
      float mn = fmaxf(mrow[r], mx);
      float c = __expf(mrow[r] - mn);
      p0[r] = __expf(s0 - mn);
      p1[r] = __expf(s1 - mn);
      lrow[r] = lrow[r] * c + redsum16(p0[r] + p1[r]);
      mrow[r] = mn;
      corr[r] = c;
    }
#pragma unroll
    for (int t = 0; t < 4; t++)
#pragma unroll
      for (int r = 0; r < 8; r++) acc[t][r] *= corr[r];

    // ---- transpose P (C/D layout -> A layout) through per-wave LDS ----
#pragma unroll
    for (int r = 0; r < 8; r++) {
      int prow = r + hi * 8;
      lds[prow * 32 + l16] = __float2bfloat16(p0[r]);
      lds[prow * 32 + 16 + l16] = __float2bfloat16(p1[r]);
    }
    asm volatile("s_wait_dscnt 0x0" ::: "memory");  // DS in-order per wave
    v16bf pa;
    {
      const bf16s* pr = &lds[l16 * 32];
      pa = combine16(*(const v4u*)(pr + hi * 16), *(const v4u*)(pr + hi * 16 + 8));
    }

    // ---- O += P * V: preload 4 V^T fragments, then chain WMMAs ----
    v16bf vf[4];
#pragma unroll
    for (int t = 0; t < 4; t++)
      vf[t] = ld_frag32(&sV[ib][(t * 16 + l16) * 32 + hi * 16]);
#pragma unroll
    for (int t = 0; t < 4; t++) acc[t] = wmma_bf16(pa, vf[t], acc[t]);
  }

  // ---- normalize + store (bf16, [B,N,D] with head interleave) ----
#pragma unroll
  for (int r = 0; r < 8; r++) lrow[r] = 1.f / lrow[r];
#pragma unroll
  for (int t = 0; t < 4; t++)
#pragma unroll
    for (int r = 0; r < 8; r++) {
      int qi = qb + r + hi * 8;
      attn_out[(size_t)(b * Nn + qi) * Dd + h * DHh + t * 16 + l16] =
          __float2bfloat16(acc[t][r] * lrow[r]);
    }
}

// ---------------- host launcher ----------------
extern "C" void kernel_launch(void* const* d_in, const int* in_sizes, int n_in,
                              void* d_out, int out_size, void* d_ws,
                              size_t ws_size, hipStream_t stream) {
  (void)in_sizes; (void)n_in; (void)out_size; (void)ws_size;
  const float* x     = (const float*)d_in[0];
  const float* gamma = (const float*)d_in[1];
  const float* dist  = (const float*)d_in[2];
  const float* ln1w  = (const float*)d_in[3];
  const float* ln1b  = (const float*)d_in[4];
  const float* qkvw  = (const float*)d_in[5];
  const float* qkvb  = (const float*)d_in[6];
  const float* projw = (const float*)d_in[7];
  const float* projb = (const float*)d_in[8];
  const float* ln2w  = (const float*)d_in[9];
  const float* ln2b  = (const float*)d_in[10];
  const float* w1    = (const float*)d_in[11];
  const float* b1    = (const float*)d_in[12];
  const float* w2    = (const float*)d_in[13];
  const float* b2    = (const float*)d_in[14];
  float* out = (float*)d_out;
  char* ws = (char*)d_ws;

  // workspace layout (bytes, all 256-aligned)
  const size_t O_QKVWT = 0;                                   // 3072*1024*2
  const size_t O_PROJWT = O_QKVWT + (size_t)D3 * Dd * 2;      // 1024*1024*2
  const size_t O_M1WT = O_PROJWT + (size_t)Dd * Dd * 2;       // 4096*1024*2
  const size_t O_M2WT = O_M1WT + (size_t)DF * Dd * 2;         // 1024*4096*2
  const size_t O_H1 = O_M2WT + (size_t)Dd * DF * 2;           // 4096*1024*2
  const size_t O_QKV = O_H1 + (size_t)Mrows * Dd * 2;         // 4096*3072*2
  const size_t O_VT = O_QKV + (size_t)Mrows * D3 * 2;         // 8 MB
  const size_t O_HID = O_QKV;  // alias: hidden (32MB) over dead qkv+vt
  const size_t O_ATTN = O_VT + (size_t)Bc * Hh * DHh * Nn * 2;
  const size_t O_X1 = O_ATTN + (size_t)Mrows * Dd * 2;        // f32 x1

  bf16s* qkvwt = (bf16s*)(ws + O_QKVWT);
  bf16s* projwt = (bf16s*)(ws + O_PROJWT);
  bf16s* m1wt = (bf16s*)(ws + O_M1WT);
  bf16s* m2wt = (bf16s*)(ws + O_M2WT);
  bf16s* h1 = (bf16s*)(ws + O_H1);      // also reused as h2
  bf16s* qkvbuf = (bf16s*)(ws + O_QKV);
  bf16s* vt = (bf16s*)(ws + O_VT);
  bf16s* hidden = (bf16s*)(ws + O_HID);
  bf16s* attn = (bf16s*)(ws + O_ATTN);
  float* x1 = (float*)(ws + O_X1);

  auto cdiv = [](size_t a, size_t b) { return (unsigned)((a + b - 1) / b); };

  // 1) weight convert + transpose to bf16 Wt[N][K]
  wt_convert_kernel<<<cdiv((size_t)Dd * D3, 256), 256, 0, stream>>>(qkvw, qkvwt, Dd, D3);
  wt_convert_kernel<<<cdiv((size_t)Dd * Dd, 256), 256, 0, stream>>>(projw, projwt, Dd, Dd);
  wt_convert_kernel<<<cdiv((size_t)Dd * DF, 256), 256, 0, stream>>>(w1, m1wt, Dd, DF);
  wt_convert_kernel<<<cdiv((size_t)DF * Dd, 256), 256, 0, stream>>>(w2, m2wt, DF, Dd);

  // 2) LN1
  ln_kernel<<<Mrows, 256, 0, stream>>>(x, ln1w, ln1b, h1);

  // 3) QKV GEMM (bf16 out + V^T epilogue)
  gemm_wmma_kernel<true, false, false, true>
      <<<dim3(D3 / 64, Mrows / 128), 256, 0, stream>>>(h1, qkvwt, qkvb, nullptr,
                                                       qkvbuf, vt, Mrows, D3, Dd);

  // 4) flash attention
  flash_attn_kernel<<<(Bc * Hh * (Nn / 16)) / 8, 256, 0, stream>>>(qkvbuf, vt,
                                                                   gamma, dist, attn);

  // 5) proj GEMM + residual -> x1 (f32)
  gemm_wmma_kernel<false, false, true, false>
      <<<dim3(Dd / 64, Mrows / 128), 256, 0, stream>>>(attn, projwt, projb, x,
                                                       x1, nullptr, Mrows, Dd, Dd);

  // 6) LN2 -> h2 (aliased onto h1)
  ln_kernel<<<Mrows, 256, 0, stream>>>(x1, ln2w, ln2b, h1);

  // 7) MLP1 + exact GELU -> hidden (bf16)
  gemm_wmma_kernel<true, true, false, false>
      <<<dim3(DF / 64, Mrows / 128), 256, 0, stream>>>(h1, m1wt, b1, nullptr,
                                                       hidden, nullptr, Mrows, DF, Dd);

  // 8) MLP2 + residual(x1) -> out (f32)
  gemm_wmma_kernel<false, false, true, false>
      <<<dim3(Dd / 64, Mrows / 128), 256, 0, stream>>>(hidden, m2wt, b2, x1,
                                                       out, nullptr, Mrows, Dd, DF);
}
